// QuantLinear_3040836846053
// MI455X (gfx1250) — compile-verified, matrix operand order checked by reading, not run on hardware
//
#include <hip/hip_runtime.h>

typedef _Float16 v16h __attribute__((ext_vector_type(16)));
typedef _Float16 v8h  __attribute__((ext_vector_type(8)));
typedef float    v8f  __attribute__((ext_vector_type(8)));
typedef float    v4f  __attribute__((ext_vector_type(4)));

constexpr int Kdim  = 4096;
constexpr int Ndim  = 4096;
constexpr int GROUP = 128;
constexpr int BM = 128, BN = 128, BK = 64;
constexpr int LDA = BK + 16;   // 80 halfs: 160B row stride, 16B aligned, spreads LDS banks
constexpr int LDB = BK + 16;

__global__ __launch_bounds__(256)
void gptq_w4a16_wmma(const float* __restrict__ x,
                     const int*   __restrict__ qweight,
                     const int*   __restrict__ qzeros,
                     const float* __restrict__ scales,
                     float*       __restrict__ out)
{
    __shared__ _Float16 sA[BM * LDA];   // x tile, f16, row-major [m][k]
    __shared__ _Float16 sB[BN * LDB];   // dequant W tile, f16, col-major [n][k]

    const int tid    = threadIdx.x;
    const int lane   = tid & 31;
    const int wave   = tid >> 5;
    const int wr     = wave >> 2;       // 0..1  -> 64 rows each
    const int wc     = wave & 3;        // 0..3  -> 32 cols each
    const int laneLo = lane & 15;
    const int laneHi = lane >> 4;

    const int blockN0 = blockIdx.x * BN;
    const int blockM0 = blockIdx.y * BM;

    // ---- x (A) load mapping: 128 rows x 64 f32 cols, 2 threads per row ----
    const int xRow = tid >> 1;
    const int xCol = (tid & 1) * 32;
    const float* xg = x + (size_t)(blockM0 + xRow) * Kdim + xCol;

    // ---- qweight (B) load mapping: 8 packed rows x 128 cols, 4 words/thread ----
    const int nLoc  = tid & 127;
    const int rQuad = (tid >> 7) * 4;           // 0 or 4
    const int nGlob = blockN0 + nLoc;

    v4f      xr[8];       // prefetched x (32 floats)
    uint32_t wq[4];       // prefetched packed weights (4 words -> 32 nibbles)
    float    sc;          // per-(group,n) scale
    int      zp;          // per-(group,n) zero point

    auto load_tile = [&](int k0) {
        const float* xp = xg + k0;
        #pragma unroll
        for (int i = 0; i < 8; ++i) xr[i] = *(const v4f*)(xp + i * 4);

        const int kw0 = (k0 >> 3) + rQuad;
        #pragma unroll
        for (int r = 0; r < 4; ++r)
            wq[r] = ((const uint32_t*)qweight)[(size_t)(kw0 + r) * Ndim + nGlob];

        const int g = k0 / GROUP;               // BK=64 divides GROUP=128 -> single group/tile
        sc = scales[(size_t)g * Ndim + nGlob];
        uint32_t zw = ((const uint32_t*)qzeros)[(size_t)g * (Ndim / 8) + (nGlob >> 3)];
        zp = (int)((zw >> ((nGlob & 7) * 4)) & 0xF);
    };

    auto store_tile = [&]() {
        // x: f32 -> f16 into sA
        _Float16* ap = sA + xRow * LDA + xCol;
        #pragma unroll
        for (int j = 0; j < 4; ++j) {
            v8h h;
            #pragma unroll
            for (int e = 0; e < 4; ++e) {
                h[e]     = (_Float16)xr[2 * j][e];
                h[4 + e] = (_Float16)xr[2 * j + 1][e];
            }
            *(v8h*)(ap + j * 8) = h;            // ds_store_b128
        }
        // qweight: dequant (q - z) * s -> f16 into sB (column-major per n)
        _Float16* bp = sB + nLoc * LDB + rQuad * 8;
        #pragma unroll
        for (int r = 0; r < 4; ++r) {
            const uint32_t w = wq[r];
            v8h h;
            #pragma unroll
            for (int i = 0; i < 8; ++i) {
                int q = (int)((w >> (4 * i)) & 0xF) - zp;
                h[i] = (_Float16)((float)q * sc);
            }
            *(v8h*)(bp + r * 8) = h;            // ds_store_b128
        }
    };

    v8f acc[4][2] = {};

    load_tile(0);
    constexpr int NT = Kdim / BK;               // 64 tiles
    for (int t = 0; t < NT; ++t) {
        store_tile();
        __syncthreads();
        if (t + 1 < NT) load_tile((t + 1) * BK);   // overlap global loads with WMMA

        #pragma unroll
        for (int kk = 0; kk < BK; kk += 32) {
            v16h aF[4], bF[2];
            #pragma unroll
            for (int i = 0; i < 4; ++i) {
                const _Float16* p = sA + (wr * 64 + i * 16 + laneLo) * LDA + kk + laneHi * 8;
                union { v16h v; v8h h[2]; } u;
                u.h[0] = *(const v8h*)p;          // ds_load_b128: K 0-7 / 8-15
                u.h[1] = *(const v8h*)(p + 16);   // ds_load_b128: K 16-23 / 24-31
                aF[i] = u.v;
            }
            #pragma unroll
            for (int j = 0; j < 2; ++j) {
                const _Float16* p = sB + (wc * 32 + j * 16 + laneLo) * LDB + kk + laneHi * 8;
                union { v16h v; v8h h[2]; } u;
                u.h[0] = *(const v8h*)p;
                u.h[1] = *(const v8h*)(p + 16);
                bF[j] = u.v;
            }
            #pragma unroll
            for (int i = 0; i < 4; ++i)
                #pragma unroll
                for (int j = 0; j < 2; ++j)
                    acc[i][j] = __builtin_amdgcn_wmma_f32_16x16x32_f16(
                        /*neg_a=*/false, aF[i], /*neg_b=*/false, bF[j],
                        /*c_mod=*/(short)0, acc[i][j],
                        /*reuse_a=*/false, /*reuse_b=*/false);
        }
        __syncthreads();
    }

    // Epilogue: C/D layout -> VGPR v = row (m%8), laneHi selects +8, lanes 0-15 = N
    #pragma unroll
    for (int i = 0; i < 4; ++i) {
        const int rowBase = blockM0 + wr * 64 + i * 16 + laneHi * 8;
        #pragma unroll
        for (int j = 0; j < 2; ++j) {
            const int col = blockN0 + wc * 32 + j * 16 + laneLo;
            #pragma unroll
            for (int v = 0; v < 8; ++v)
                out[(size_t)(rowBase + v) * Ndim + col] = acc[i][j][v];
        }
    }
}

extern "C" void kernel_launch(void* const* d_in, const int* in_sizes, int n_in,
                              void* d_out, int out_size, void* d_ws, size_t ws_size,
                              hipStream_t stream) {
    const float* x       = (const float*)d_in[0];
    const int*   qweight = (const int*)d_in[1];
    const int*   qzeros  = (const int*)d_in[2];
    const float* scales  = (const float*)d_in[3];
    // d_in[4] = g_idx; implicit as k / GROUP (matches reference setup)
    float* out = (float*)d_out;

    const int tokens = in_sizes[0] / Kdim;      // 8192
    dim3 grid(Ndim / BN, tokens / BM);          // (32, 64)
    gptq_w4a16_wmma<<<grid, 256, 0, stream>>>(x, qweight, qzeros, scales, out);
}
